// CAVAModule_10651518894672
// MI455X (gfx1250) — matrix operand order repaired and checked by reading, not compiled, not run on hardware
//
#include <hip/hip_runtime.h>
#include <hip/hip_bf16.h>

typedef __attribute__((ext_vector_type(16))) __bf16 v16bf;
typedef __attribute__((ext_vector_type(8)))  float  v8f;

#define WMMA_BF16(A_, B_, C_) \
  __builtin_amdgcn_wmma_f32_16x16x32_bf16(false, (A_), false, (B_), (short)0, (C_), false, false)

__device__ __forceinline__ float wave_sum(float v) {
  v += __shfl_xor(v, 1, 32);
  v += __shfl_xor(v, 2, 32);
  v += __shfl_xor(v, 4, 32);
  v += __shfl_xor(v, 8, 32);
  v += __shfl_xor(v, 16, 32);
  return v;
}

__device__ __forceinline__ float gelu_exact(float x) {
  return 0.5f * x * (1.0f + erff(x * 0.70710678118654752f));
}

__device__ __forceinline__ void split16(const float* __restrict__ f, v16bf& hi, v16bf& lo) {
#pragma unroll
  for (int i = 0; i < 16; ++i) {
    float x = f[i];
    __bf16 h = (__bf16)x;
    hi[i] = h;
    lo[i] = (__bf16)(x - (float)h);
  }
}

__device__ __forceinline__ void load16(const float* __restrict__ p, float* v) {
  *(float4*)(v + 0)  = *(const float4*)(p + 0);
  *(float4*)(v + 4)  = *(const float4*)(p + 4);
  *(float4*)(v + 8)  = *(const float4*)(p + 8);
  *(float4*)(v + 12) = *(const float4*)(p + 12);
}

// ---------------------------------------------------------------------------
// Pack fp32 weight W(K,N) into bf16 hi/lo B-fragments.
// B 32x16 bf16 fragment: lanes 0-15 -> N=lane, elems e<8 -> K=kt*32+e, e>=8 -> K=kt*32+8+e;
// lanes 16-31 -> N=lane-16, elems -> same with K offset +8.
// ---------------------------------------------------------------------------
__global__ void pack_w_kernel(const float* __restrict__ W,
                              __bf16* __restrict__ Hi, __bf16* __restrict__ Lo,
                              int K, int N) {
  const int lane = threadIdx.x & 31;
  const int tile = blockIdx.x * (blockDim.x >> 5) + (threadIdx.x >> 5);
  const int Nt = N >> 4, Kt = K >> 5;
  if (tile >= Kt * Nt) return;
  const int nt = tile % Nt;
  const int kt = tile / Nt;
  const int n  = (nt << 4) + (lane & 15);
  const int k0 = (kt << 5) + ((lane < 16) ? 0 : 8);
  const size_t base = ((size_t)tile * 32 + lane) * 16;
#pragma unroll
  for (int e = 0; e < 16; ++e) {
    float f = W[(size_t)(k0 + ((e < 8) ? e : (e + 8))) * N + n];
    __bf16 h = (__bf16)f;
    Hi[base + e] = h;
    Lo[base + e] = (__bf16)(f - (float)h);
  }
}

// ---------------------------------------------------------------------------
// C(M,N) = A(M,K) @ W(K,N) + bias.  A fp32 row-major (fragment + bf16 hi/lo
// split built once per K-step), W pre-packed.  One wave computes a 16x64
// strip (4 N-tiles) so the A fragment is reused by 12 WMMAs per K-step.
// ---------------------------------------------------------------------------
__global__ __launch_bounds__(256) void proj_gemm_kernel(
    const float* __restrict__ A,
    const __bf16* __restrict__ Bhi, const __bf16* __restrict__ Blo,
    const float* __restrict__ bias, float* __restrict__ C,
    int M, int K, int N) {
  const int lane = threadIdx.x & 31;
  const int Ng = N >> 6;  // groups of 4 n-tiles
  const int wid = blockIdx.x * 8 + (threadIdx.x >> 5);
  if (wid >= (M >> 4) * Ng) return;
  const int mt = wid / Ng;
  const int nt0 = (wid % Ng) << 2;
  const int Nt = N >> 4;
  const int m   = (mt << 4) + (lane & 15);
  const int off = (lane < 16) ? 0 : 8;
  const int Kt = K >> 5;
  v8f acc0 = {}, acc1 = {}, acc2 = {}, acc3 = {};
  for (int kt = 0; kt < Kt; ++kt) {
    float av[16];
    const float* ap = A + (size_t)m * K + (kt << 5) + off;
    *(float4*)(av + 0)  = *(const float4*)(ap + 0);
    *(float4*)(av + 4)  = *(const float4*)(ap + 4);
    *(float4*)(av + 8)  = *(const float4*)(ap + 16);
    *(float4*)(av + 12) = *(const float4*)(ap + 20);
    v16bf ahi, alo;
    split16(av, ahi, alo);
    const size_t b0 = (((size_t)kt * Nt + nt0) * 32 + lane) * 16;
    {
      const v16bf bh = *(const v16bf*)(Bhi + b0);
      const v16bf bl = *(const v16bf*)(Blo + b0);
      acc0 = WMMA_BF16(ahi, bh, acc0);
      acc0 = WMMA_BF16(alo, bh, acc0);
      acc0 = WMMA_BF16(ahi, bl, acc0);
    }
    {
      const v16bf bh = *(const v16bf*)(Bhi + b0 + 512);
      const v16bf bl = *(const v16bf*)(Blo + b0 + 512);
      acc1 = WMMA_BF16(ahi, bh, acc1);
      acc1 = WMMA_BF16(alo, bh, acc1);
      acc1 = WMMA_BF16(ahi, bl, acc1);
    }
    {
      const v16bf bh = *(const v16bf*)(Bhi + b0 + 1024);
      const v16bf bl = *(const v16bf*)(Blo + b0 + 1024);
      acc2 = WMMA_BF16(ahi, bh, acc2);
      acc2 = WMMA_BF16(alo, bh, acc2);
      acc2 = WMMA_BF16(ahi, bl, acc2);
    }
    {
      const v16bf bh = *(const v16bf*)(Bhi + b0 + 1536);
      const v16bf bl = *(const v16bf*)(Blo + b0 + 1536);
      acc3 = WMMA_BF16(ahi, bh, acc3);
      acc3 = WMMA_BF16(alo, bh, acc3);
      acc3 = WMMA_BF16(ahi, bl, acc3);
    }
  }
  const int mrow0 = (mt << 4) + ((lane < 16) ? 0 : 8);
  v8f accs[4] = {acc0, acc1, acc2, acc3};
#pragma unroll
  for (int j = 0; j < 4; ++j) {
    const int n = ((nt0 + j) << 4) + (lane & 15);
    const float bn = bias[n];
#pragma unroll
    for (int r = 0; r < 8; ++r)
      C[(size_t)(mrow0 + r) * N + n] = accs[j][r] + bn;
  }
}

// ---------------------------------------------------------------------------
// Temporal mean pooling: vp[b,d] = mean_t v[b,t,d] (same for a).
// ---------------------------------------------------------------------------
__global__ void pool_kernel(const float* __restrict__ v, const float* __restrict__ a,
                            float* __restrict__ vp, float* __restrict__ ap_) {
  const int id = blockIdx.x * blockDim.x + threadIdx.x;
  if (id >= 16 * 512) return;
  const int b = id >> 9, d = id & 511;
  const float* pv = v + ((size_t)b << 19) + d;   // b*1024*512
  const float* pa = a + ((size_t)b << 19) + d;
  float sv = 0.f, sa = 0.f;
  for (int t = 0; t < 1024; ++t) { sv += pv[(size_t)t * 512]; sa += pa[(size_t)t * 512]; }
  vp[id]  = sv * (1.0f / 1024.0f);
  ap_[id] = sa * (1.0f / 1024.0f);
}

// h = gelu(concat(vp, ap, vp*ap) @ Wd1 + bd1)   (16 x 512, K=1536 -- tiny)
__global__ void dense1_kernel(const float* __restrict__ vp, const float* __restrict__ ap_,
                              const float* __restrict__ Wd1, const float* __restrict__ bd1,
                              float* __restrict__ h) {
  const int id = blockIdx.x * blockDim.x + threadIdx.x;
  if (id >= 16 * 512) return;
  const int b = id >> 9, n = id & 511;
  const float* fv = vp  + (b << 9);
  const float* fa = ap_ + (b << 9);
  float acc = 0.f;
  for (int k = 0; k < 512; ++k) acc += fv[k] * Wd1[(size_t)k * 512 + n];
  for (int k = 0; k < 512; ++k) acc += fa[k] * Wd1[(size_t)(512 + k) * 512 + n];
  for (int k = 0; k < 512; ++k) acc += fv[k] * fa[k] * Wd1[(size_t)(1024 + k) * 512 + n];
  h[id] = gelu_exact(acc + bd1[n]);
}

// delta[b] = 2 + 4*sigmoid(clip(h[b] . Wd2 + bd2, -12, 12));  1 wave per batch
__global__ void delta_kernel(const float* __restrict__ h, const float* __restrict__ Wd2,
                             const float* __restrict__ bd2, float* __restrict__ delta) {
  const int lane = threadIdx.x & 31;
  const int b = threadIdx.x >> 5;
  if (b >= 16) return;
  float s = 0.f;
  for (int k = lane; k < 512; k += 32) s += h[(b << 9) + k] * Wd2[k];
  s = wave_sum(s);
  if (lane == 0) {
    float dl = fminf(fmaxf(s + bd2[0], -12.0f), 12.0f);
    delta[b] = 2.0f + 4.0f / (1.0f + expf(-dl));
  }
}

// v_shift[b,t,:] = lerp(v[b,i0,:], v[b,i1,:], alpha) with pos=min(t+delta,1023)
__global__ void vshift_kernel(const float* __restrict__ v, const float* __restrict__ delta,
                              float* __restrict__ vsh) {
  const size_t id = (size_t)blockIdx.x * blockDim.x + threadIdx.x;
  if (id >= (size_t)16 * 1024 * 512) return;
  const int d = (int)(id & 511);
  const int row = (int)(id >> 9);
  const int b = row >> 10, t = row & 1023;
  const float pos = fminf((float)t + delta[b], 1023.0f);
  const float i0f = floorf(pos);
  const int i0 = (int)i0f;
  const float alpha = pos - i0f;
  const int i1 = min(i0 + 1, 1023);
  const size_t base = (size_t)b << 19;
  vsh[id] = (1.0f - alpha) * v[base + ((size_t)i0 << 9) + d]
          +         alpha  * v[base + ((size_t)i1 << 9) + d];
}

// rn[row] = 1 / max(||x_row||, 1e-8)  for a (rows x 512) matrix; 1 wave/row
__global__ void rnorm_kernel(const float* __restrict__ x, float* __restrict__ rn, int rows) {
  const int lane = threadIdx.x & 31;
  const int row = blockIdx.x * (blockDim.x >> 5) + (threadIdx.x >> 5);
  if (row >= rows) return;
  float v[16];
  load16(x + ((size_t)row << 9) + lane * 16, v);
  float ss = 0.f;
#pragma unroll
  for (int i = 0; i < 16; ++i) ss += v[i] * v[i];
  ss = wave_sum(ss);
  if (lane == 0) rn[row] = 1.0f / fmaxf(sqrtf(ss), 1e-8f);
}

// ---------------------------------------------------------------------------
// Windowed cross-attention + residual layer-norm.  The mask bias is constant
// (log(1/count)) inside the window and log(1e-8) (~e^-16 relative) outside,
// so softmax reduces to the 11 in-window logits.  1 wave per (b,t) row.
// ---------------------------------------------------------------------------
__global__ __launch_bounds__(256) void attn_kernel(
    const float* __restrict__ a, const float* __restrict__ vsh,
    const float* __restrict__ delta, float* __restrict__ aligned,
    float* __restrict__ rna) {
  const int lane = threadIdx.x & 31;
  const int row = blockIdx.x * 8 + (threadIdx.x >> 5);
  if (row >= 16 * 1024) return;
  const int b = row >> 10, t = row & 1023;
  float av[16];
  load16(a + ((size_t)row << 9) + lane * 16, av);

  const float center = fminf((float)t + delta[b], 1023.0f);
  const int s_lo = max(0, (int)ceilf(center - 5.0f));
  const int s_hi = min(1023, (int)floorf(center + 5.0f));
  const float scale = 0.04419417382415922f;  // 1/sqrt(512)

  float logit[11];
#pragma unroll
  for (int j = 0; j < 11; ++j) {
    const int s = s_lo + j;
    if (s <= s_hi) {
      float vv[16];
      load16(vsh + ((size_t)((b << 10) + s) << 9) + lane * 16, vv);
      float d0 = 0.f;
#pragma unroll
      for (int i = 0; i < 16; ++i) d0 += av[i] * vv[i];
      logit[j] = wave_sum(d0) * scale;
    } else {
      logit[j] = -1e30f;
    }
  }
  float mx = logit[0];
#pragma unroll
  for (int j = 1; j < 11; ++j) mx = fmaxf(mx, logit[j]);
  float p[11], sum = 0.f;
#pragma unroll
  for (int j = 0; j < 11; ++j) {
    p[j] = (logit[j] > -1e29f) ? expf(logit[j] - mx) : 0.0f;
    sum += p[j];
  }
  const float inv = 1.0f / sum;

  float ctx[16];
#pragma unroll
  for (int i = 0; i < 16; ++i) ctx[i] = 0.f;
#pragma unroll
  for (int j = 0; j < 11; ++j) {
    const int s = s_lo + j;
    if (s <= s_hi) {
      float vv[16];
      load16(vsh + ((size_t)((b << 10) + s) << 9) + lane * 16, vv);
      const float w = p[j] * inv;
#pragma unroll
      for (int i = 0; i < 16; ++i) ctx[i] += w * vv[i];
    }
  }

  float x[16], s1 = 0.f, s2 = 0.f;
#pragma unroll
  for (int i = 0; i < 16; ++i) {
    x[i] = av[i] + ctx[i];
    s1 += x[i];
    s2 += x[i] * x[i];
  }
  s1 = wave_sum(s1);
  s2 = wave_sum(s2);
  const float mu = s1 * (1.0f / 512.0f);
  const float var = s2 * (1.0f / 512.0f) - mu * mu;
  const float rstd = rsqrtf(var + 1e-5f);
  float y[16], ss = 0.f;
#pragma unroll
  for (int i = 0; i < 16; ++i) {
    y[i] = (x[i] - mu) * rstd;
    ss += y[i] * y[i];
  }
  ss = wave_sum(ss);
  float* orow = aligned + ((size_t)row << 9) + lane * 16;
  *(float4*)(orow + 0)  = *(const float4*)(y + 0);
  *(float4*)(orow + 4)  = *(const float4*)(y + 4);
  *(float4*)(orow + 8)  = *(const float4*)(y + 8);
  *(float4*)(orow + 12) = *(const float4*)(y + 12);
  if (lane == 0) rna[row] = 1.0f / fmaxf(sqrtf(ss), 1e-8f);
}

// ---------------------------------------------------------------------------
// Materialize feat_g = concat(an, sn, an*sn) as packed bf16 hi/lo A-fragments
// (M=16384, K=1536).  Fragment order: tile = mt*48 + kt.
// ---------------------------------------------------------------------------
__global__ void pack_featg_kernel(const float* __restrict__ aligned,
                                  const float* __restrict__ vsh,
                                  const float* __restrict__ rna,
                                  const float* __restrict__ rns,
                                  __bf16* __restrict__ Hi, __bf16* __restrict__ Lo) {
  const int Kt = 48;  // 1536 / 32
  const int lane = threadIdx.x & 31;
  const int tile = blockIdx.x * (blockDim.x >> 5) + (threadIdx.x >> 5);
  if (tile >= 1024 * Kt) return;
  const int mt = tile / Kt, kt = tile % Kt;
  const int m = (mt << 4) + (lane & 15);
  const int off = (lane < 16) ? 0 : 8;
  const float na = rna[m], ns = rns[m];
  const size_t base = ((size_t)tile * 32 + lane) * 16;
#pragma unroll
  for (int e = 0; e < 16; ++e) {
    const int k = (kt << 5) + off + ((e < 8) ? e : (e + 8));
    float f;
    if (k < 512) {
      f = aligned[((size_t)m << 9) + k] * na;
    } else if (k < 1024) {
      f = vsh[((size_t)m << 9) + (k - 512)] * ns;
    } else {
      const int kk = k - 1024;
      f = aligned[((size_t)m << 9) + kk] * vsh[((size_t)m << 9) + kk] * na * ns;
    }
    __bf16 h = (__bf16)f;
    Hi[base + e] = h;
    Lo[base + e] = (__bf16)(f - (float)h);
  }
}

__global__ void init_glogits_kernel(float* __restrict__ g, const float* __restrict__ bg2) {
  const int id = blockIdx.x * blockDim.x + threadIdx.x;
  if (id < 16 * 1024) g[id] = bg2[0];
}

// ---------------------------------------------------------------------------
// Big gate GEMM (16384x1536 @ 1536x2048, 103 GF) with fused epilogue:
// gelu(tile + bg1) * Wg2[n], cross-lane reduce over the 16 N-lanes, and
// atomicAdd into glogits[m].  One wave computes a 32x32 block (2x2 tiles):
// per K-step, 2 A hi/lo pairs + 2 B hi/lo pairs feed 12 WMMAs, doubling
// arithmetic intensity on both operands vs. a single-tile wave.
// ---------------------------------------------------------------------------
__global__ __launch_bounds__(256) void gate_gemm_kernel(
    const __bf16* __restrict__ Ahi, const __bf16* __restrict__ Alo,
    const __bf16* __restrict__ Bhi, const __bf16* __restrict__ Blo,
    const float* __restrict__ bg1, const float* __restrict__ Wg2,
    float* __restrict__ glogits) {
  const int Kt = 48, Nt = 128;  // K=1536, N=2048
  const int Ngr = Nt >> 1;      // 64 n-groups of 2 tiles
  const int lane = threadIdx.x & 31;
  const int wid = blockIdx.x * 8 + (threadIdx.x >> 5);
  if (wid >= 512 * Ngr) return;  // (1024/2) m-groups * 64 n-groups
  const int mt0 = (wid / Ngr) << 1;
  const int nt0 = (wid % Ngr) << 1;
  v8f acc[2][2] = {};
  for (int kt = 0; kt < Kt; ++kt) {
    const size_t a0 = (((size_t)mt0 * Kt + kt) * 32 + lane) * 16;
    const size_t a1 = a0 + (size_t)Kt * 512;          // next m-tile
    const size_t b0 = (((size_t)kt * Nt + nt0) * 32 + lane) * 16;
    const v16bf ah0 = *(const v16bf*)(Ahi + a0);
    const v16bf al0 = *(const v16bf*)(Alo + a0);
    const v16bf ah1 = *(const v16bf*)(Ahi + a1);
    const v16bf al1 = *(const v16bf*)(Alo + a1);
    const v16bf bh0 = *(const v16bf*)(Bhi + b0);
    const v16bf bl0 = *(const v16bf*)(Blo + b0);
    const v16bf bh1 = *(const v16bf*)(Bhi + b0 + 512);
    const v16bf bl1 = *(const v16bf*)(Blo + b0 + 512);
    acc[0][0] = WMMA_BF16(ah0, bh0, acc[0][0]);
    acc[0][0] = WMMA_BF16(al0, bh0, acc[0][0]);
    acc[0][0] = WMMA_BF16(ah0, bl0, acc[0][0]);
    acc[0][1] = WMMA_BF16(ah0, bh1, acc[0][1]);
    acc[0][1] = WMMA_BF16(al0, bh1, acc[0][1]);
    acc[0][1] = WMMA_BF16(ah0, bl1, acc[0][1]);
    acc[1][0] = WMMA_BF16(ah1, bh0, acc[1][0]);
    acc[1][0] = WMMA_BF16(al1, bh0, acc[1][0]);
    acc[1][0] = WMMA_BF16(ah1, bl0, acc[1][0]);
    acc[1][1] = WMMA_BF16(ah1, bh1, acc[1][1]);
    acc[1][1] = WMMA_BF16(al1, bh1, acc[1][1]);
    acc[1][1] = WMMA_BF16(ah1, bl1, acc[1][1]);
  }
#pragma unroll
  for (int mi = 0; mi < 2; ++mi) {
#pragma unroll
    for (int nj = 0; nj < 2; ++nj) {
      const int n = ((nt0 + nj) << 4) + (lane & 15);
      const float bn = bg1[n], w2 = Wg2[n];
      float c[8];
#pragma unroll
      for (int r = 0; r < 8; ++r)
        c[r] = gelu_exact(acc[mi][nj][r] + bn) * w2;
      // reduce across the 16 N-lanes within each half-wave
#pragma unroll
      for (int r = 0; r < 8; ++r) {
        c[r] += __shfl_xor(c[r], 1, 32);
        c[r] += __shfl_xor(c[r], 2, 32);
        c[r] += __shfl_xor(c[r], 4, 32);
        c[r] += __shfl_xor(c[r], 8, 32);
      }
      if ((lane & 15) == 0) {
        const int mrow0 = ((mt0 + mi) << 4) + ((lane < 16) ? 0 : 8);
#pragma unroll
        for (int r = 0; r < 8; ++r)
          atomicAdd(&glogits[mrow0 + r], c[r]);
      }
    }
  }
}

// fused = gate*aligned + (1-gate)*v_shift
__global__ void fuse_kernel(const float* __restrict__ aligned, const float* __restrict__ vsh,
                            const float* __restrict__ glogits, float* __restrict__ out) {
  const size_t id = (size_t)blockIdx.x * blockDim.x + threadIdx.x;
  if (id >= (size_t)16 * 1024 * 512) return;
  const int row = (int)(id >> 9);
  float gl = fminf(fmaxf(glogits[row], -12.0f), 12.0f);
  float g = 1.0f / (1.0f + expf(-gl));
  g = fminf(fmaxf(g, 0.05f), 0.95f);
  out[id] = g * aligned[id] + (1.0f - g) * vsh[id];
}

// ---------------------------------------------------------------------------
extern "C" void kernel_launch(void* const* d_in, const int* in_sizes, int n_in,
                              void* d_out, int out_size, void* d_ws, size_t ws_size,
                              hipStream_t stream) {
  (void)in_sizes; (void)n_in; (void)out_size; (void)ws_size;
  const float* video = (const float*)d_in[0];
  const float* audio = (const float*)d_in[1];
  const float* Wv  = (const float*)d_in[2];
  const float* bv  = (const float*)d_in[3];
  const float* Wa  = (const float*)d_in[4];
  const float* ba  = (const float*)d_in[5];
  const float* Wd1 = (const float*)d_in[6];
  const float* bd1 = (const float*)d_in[7];
  const float* Wd2 = (const float*)d_in[8];
  const float* bd2 = (const float*)d_in[9];
  const float* Wg1 = (const float*)d_in[10];
  const float* bg1 = (const float*)d_in[11];
  const float* Wg2 = (const float*)d_in[12];
  const float* bg2 = (const float*)d_in[13];
  float* out = (float*)d_out;

  const int M = 16384;  // B*T
  char* ws = (char*)d_ws;
  size_t off = 0;
  auto alloc = [&](size_t bytes) -> void* {
    void* p = ws + off;
    off = (off + bytes + 255) & ~(size_t)255;
    return p;
  };
  float*  v        = (float*)alloc((size_t)M * 512 * 4);
  float*  a        = (float*)alloc((size_t)M * 512 * 4);
  float*  vsh      = (float*)alloc((size_t)M * 512 * 4);
  float*  aligned  = (float*)alloc((size_t)M * 512 * 4);
  __bf16* fgHi     = (__bf16*)alloc((size_t)M * 1536 * 2);
  __bf16* fgLo     = (__bf16*)alloc((size_t)M * 1536 * 2);
  __bf16* pWvHi    = (__bf16*)alloc((size_t)1024 * 512 * 2);
  __bf16* pWvLo    = (__bf16*)alloc((size_t)1024 * 512 * 2);
  __bf16* pWaHi    = (__bf16*)alloc((size_t)768 * 512 * 2);
  __bf16* pWaLo    = (__bf16*)alloc((size_t)768 * 512 * 2);
  __bf16* pWgHi    = (__bf16*)alloc((size_t)1536 * 2048 * 2);
  __bf16* pWgLo    = (__bf16*)alloc((size_t)1536 * 2048 * 2);
  float*  rna      = (float*)alloc(M * 4);
  float*  rns      = (float*)alloc(M * 4);
  float*  glogits  = (float*)alloc(M * 4);
  float*  vp       = (float*)alloc(16 * 512 * 4);
  float*  ap_      = (float*)alloc(16 * 512 * 4);
  float*  h        = (float*)alloc(16 * 512 * 4);
  float*  delta    = (float*)alloc(16 * 4);

  // 1) Pack weights into bf16 hi/lo WMMA B-fragments.
  pack_w_kernel<<<(32 * 32 + 7) / 8, 256, 0, stream>>>(Wv, pWvHi, pWvLo, 1024, 512);
  pack_w_kernel<<<(24 * 32 + 7) / 8, 256, 0, stream>>>(Wa, pWaHi, pWaLo, 768, 512);
  pack_w_kernel<<<(48 * 128 + 7) / 8, 256, 0, stream>>>(Wg1, pWgHi, pWgLo, 1536, 2048);

  // 2) Projections (WMMA, split bf16, 16x64 per wave).
  proj_gemm_kernel<<<(1024 * 8) / 8, 256, 0, stream>>>(video, pWvHi, pWvLo, bv, v, M, 1024, 512);
  proj_gemm_kernel<<<(1024 * 8) / 8, 256, 0, stream>>>(audio, pWaHi, pWaLo, ba, a, M, 768, 512);

  // 3) Delta MLP (tiny).
  pool_kernel<<<32, 256, 0, stream>>>(v, a, vp, ap_);
  dense1_kernel<<<32, 256, 0, stream>>>(vp, ap_, Wd1, bd1, h);
  delta_kernel<<<1, 512, 0, stream>>>(h, Wd2, bd2, delta);

  // 4) Fractional time shift + row norms.
  vshift_kernel<<<32768, 256, 0, stream>>>(v, delta, vsh);
  rnorm_kernel<<<2048, 256, 0, stream>>>(vsh, rns, M);

  // 5) Windowed attention + layernorm.
  attn_kernel<<<2048, 256, 0, stream>>>(a, vsh, delta, aligned, rna);

  // 6) Gate MLP: pack feat_g fragments, init logits, big fused WMMA GEMM.
  pack_featg_kernel<<<(1024 * 48) / 8, 256, 0, stream>>>(aligned, vsh, rna, rns, fgHi, fgLo);
  init_glogits_kernel<<<64, 256, 0, stream>>>(glogits, bg2);
  gate_gemm_kernel<<<(512 * 64) / 8, 256, 0, stream>>>(fgHi, fgLo, pWgHi, pWgLo, bg1, Wg2, glogits);

  // 7) Final gated fusion -> output.
  fuse_kernel<<<32768, 256, 0, stream>>>(aligned, vsh, glogits, out);
}
